// SSMEvaluator_33406255629019
// MI455X (gfx1250) — matrix operand order repaired and checked by reading, not compiled, not run on hardware
//
#include <hip/hip_runtime.h>
#include <hip/hip_bf16.h>
#include <cstdint>

// ---------------- problem constants ----------------
#define S_   128
#define B_   64
#define NS_  128
#define NC_  8
#define D_   32
#define Z_   64
#define H_   128
#define HI_  128
#define NF_  2
#define N_   (NS_*B_)          // 8192 independent rows
#define LOG2PI_ 1.8378770664093453f

// ---------------- packed-weight fragment offsets (in halves; 512 halves per 32x16 frag) ----
#define OFF_Wxh 0        // KT=1 NT=8  (8 frags)
#define OFF_Wzh 4096     // KT=2 NT=8  (16)
#define OFF_Wg  12288    // KT=4 NT=4  (16)
#define OFF_Wp  20480    // KT=4 NT=4  (16)
#define OFF_Wz  28672    // KT=2 NT=4  (8)
#define OFF_Wsc 32768    // KT=2 NT=4  (8)
#define OFF_W1  36864    // per-flow stride 8192 (KT=2 NT=8)
#define OFF_Wm  53248    // per-flow stride 8192 (KT=4 NT=4)
#define OFF_Ws  69632    // per-flow stride 8192 (KT=4 NT=4)
#define OFF_We1 86016    // KT=2 NT=8 (16)
#define OFF_We2 94208    // KT=4 NT=2 (8)
#define W_HALFS 98304    // 192 KB of f16 weights

// bias layout in LDS (floats)
#define BHo  0
#define BGo  128
#define BPo  192
#define BZo  256
#define BSCo 320
#define B1o  384   // + f*128
#define BMo  640   // + f*64
#define BSo  768   // + f*64
#define BE1o 896
#define BE2o 1024
#define BIAS_FLOATS 1088

#define WAVES 4
#define SCR_HALFS_PER_WAVE 4096          // zbuf 16x64 + hbuf 16x128 + pbuf 16x64
#define SMEM_BYTES (W_HALFS*2 + BIAS_FLOATS*4 + WAVES*SCR_HALFS_PER_WAVE*2) // 233,728 B < 320 KB

typedef __attribute__((ext_vector_type(16))) _Float16 v16h;
typedef __attribute__((ext_vector_type(8)))  _Float16 v8h;
typedef __attribute__((ext_vector_type(8)))  float    v8f;

// ---------------- device helpers ----------------
__device__ __forceinline__ v8f wmma_f16(v16h a, v16h b, v8f c) {
  return __builtin_amdgcn_wmma_f32_16x16x32_f16(false, a, false, b, (short)0, c, false, false);
}

// A fragment (16x32 f16 tile, row-major in LDS): per ISA layout, each lane reads two
// contiguous 8-half chunks: rows m=lane&15, K = kb..kb+7 and 16+kb..16+kb+7, kb=(lane>=16)*8
__device__ __forceinline__ v16h ldsA(const _Float16* p, int stride, int lane) {
  const int m = lane & 15, kbh = (lane >> 4) << 3;
  const _Float16* r = p + m*stride + kbh;
  v8h c0 = *(const v8h*)r;
  v8h c1 = *(const v8h*)(r + 16);
  v16h a;
#pragma unroll
  for (int i = 0; i < 8; ++i) { a[i] = c0[i]; a[i+8] = c1[i]; }
  return a;
}

// B fragment pre-packed lane-major: 16 contiguous halves per lane (32B, aligned)
__device__ __forceinline__ v16h ldB(const _Float16* frag, int lane) {
  return *(const v16h*)(frag + lane*16);
}

__device__ __forceinline__ v8f bias8(const float* b, int lane) {
  float v = b[lane & 15];
  v8f a;
#pragma unroll
  for (int r = 0; r < 8; ++r) a[r] = v;
  return a;
}

// store C/D tile (f32 regs) as f16 row-major: lane holds col n=lane&15, rows mb..mb+7
__device__ __forceinline__ void storeD(_Float16* p, int stride, int lane, v8f a, bool relu) {
  const int n = lane & 15, mbv = (lane >> 4) << 3;
#pragma unroll
  for (int r = 0; r < 8; ++r) {
    float v = a[r];
    if (relu) v = v > 0.f ? v : 0.f;
    p[(mbv + r)*stride + n] = (_Float16)v;
  }
}

__device__ __forceinline__ float sigf(float x) { return 1.f/(1.f + __expf(-x)); }
__device__ __forceinline__ float softplusf(float x) {
  float ax = fabsf(x);
  return fmaxf(x, 0.f) + __logf(1.f + __expf(-ax));
}

// ---------------- prep: pack W (KxN f32, optional IAF mask) into WMMA-B fragment order ----
__global__ void pack_weights(const float* __restrict__ W, int K, int N,
                             _Float16* __restrict__ dst, int maskMode) {
  const int frag = blockIdx.x;
  const int KT = K >> 5;
  const int kt = frag % KT, nt = frag / KT;
  const int tid = threadIdx.x;          // 512 threads: lane-major
  const int l = tid >> 4, e = tid & 15;
  const int n = (nt << 4) + (l & 15);
  const int k = (kt << 5) + ((e >> 3) << 4) + ((l >> 4) << 3) + (e & 7);   // B layout: K = kb..kb+7, 16+kb..16+kb+7, kb=(lane>=16)*8
  float v = W[k*N + n];
  if (maskMode == 1)      { if (!((n % (Z_-1)) >= k)) v = 0.f; } // MASK1: Z x HI
  else if (maskMode == 2) { if (!(n > (k % (Z_-1))))  v = 0.f; } // MASK2: HI x Z
  dst[(size_t)frag*512 + tid] = (_Float16)v;
}

// ---------------- main persistent-scan kernel ----------------
__global__ __launch_bounds__(WAVES*32, 1)
void ssm_kernel(const float* __restrict__ x, const float* __restrict__ eps,
                const float* __restrict__ u, const _Float16* __restrict__ pw,
                const float* __restrict__ bh, const float* __restrict__ bg,
                const float* __restrict__ bp, const float* __restrict__ bz,
                const float* __restrict__ bsc, const float* __restrict__ b1,
                const float* __restrict__ bm, const float* __restrict__ bs,
                const float* __restrict__ be1, const float* __restrict__ be2,
                float* __restrict__ o_samp, float* __restrict__ o_frac,
                float* __restrict__ o_xlp, float* __restrict__ o_zlp) {
  extern __shared__ char smem[];
  _Float16* Wl = (_Float16*)smem;                       // 192 KB packed weights
  float*    bl = (float*)(smem + W_HALFS*2);            // biases
  _Float16* scr = (_Float16*)(smem + W_HALFS*2 + BIAS_FLOATS*4);

  const int tid = threadIdx.x;
  { // cooperative LDS fill: weights (b128 copies) + biases
    const uint4* src = (const uint4*)pw; uint4* dst = (uint4*)Wl;
    for (int i = tid; i < (W_HALFS*2)/16; i += WAVES*32) dst[i] = src[i];
    for (int i = tid; i < 128; i += WAVES*32) bl[BHo  + i] = bh[i];
    for (int i = tid; i <  64; i += WAVES*32) bl[BGo  + i] = bg[i];
    for (int i = tid; i <  64; i += WAVES*32) bl[BPo  + i] = bp[i];
    for (int i = tid; i <  64; i += WAVES*32) bl[BZo  + i] = bz[i];
    for (int i = tid; i <  64; i += WAVES*32) bl[BSCo + i] = bsc[i];
    for (int i = tid; i < 256; i += WAVES*32) bl[B1o  + i] = b1[i];
    for (int i = tid; i < 128; i += WAVES*32) bl[BMo  + i] = bm[i];
    for (int i = tid; i < 128; i += WAVES*32) bl[BSo  + i] = bs[i];
    for (int i = tid; i < 128; i += WAVES*32) bl[BE1o + i] = be1[i];
    for (int i = tid; i <  32; i += WAVES*32) bl[BE2o + i] = be2[i];
  }
  __syncthreads();

  const int wave = tid >> 5, lane = tid & 31;
  _Float16* zbuf = scr + wave*SCR_HALFS_PER_WAVE;   // 16 x Z   f16
  _Float16* hbuf = zbuf + 1024;                     // 16 x H   f16
  _Float16* pbuf = hbuf + 2048;                     // 16 x Z   f16
  const int rowBase = (blockIdx.x*WAVES + wave) * 16;
  const int m   = lane & 15;
  const int kbh = (lane >> 4) << 3;   // A-layout K base
  const int mb  = (lane >> 4) << 3;   // D-layout row base
  const int nD  = lane & 15;          // D-layout col within tile

  for (int i = lane; i < 1024; i += 32) zbuf[i] = (_Float16)0.f;  // z0 = 0

#pragma unroll 1
  for (int t = 0; t < S_; ++t) {
    // ---- A fragments: x_t row tile (rows repeat mod B) and z_prev
    v16h ax;
    {
      const float* xp = x + ((size_t)t*B_ + ((rowBase + m) & (B_-1)))*D_;
#pragma unroll
      for (int e = 0; e < 16; ++e) ax[e] = (_Float16)xp[((e>>3)<<4) + kbh + (e&7)];
    }
    v16h az0 = ldsA(zbuf,      Z_, lane);
    v16h az1 = ldsA(zbuf + 32, Z_, lane);

    // ---- h = relu(x@Wxh + z@Wzh + bh) -> hbuf
#pragma unroll
    for (int nt = 0; nt < 8; ++nt) {
      v8f acc = bias8(bl + BHo + nt*16, lane);
      acc = wmma_f16(ax,  ldB(Wl + OFF_Wxh +  nt       *512, lane), acc);
      acc = wmma_f16(az0, ldB(Wl + OFF_Wzh + (nt*2+0)  *512, lane), acc);
      acc = wmma_f16(az1, ldB(Wl + OFF_Wzh + (nt*2+1)  *512, lane), acc);
      storeD(hbuf + nt*16, H_, lane, acc, true);
    }

    // ---- gate/prop/zWz -> loc ; relu(prop) -> pbuf
    v16h ah0 = ldsA(hbuf,      H_, lane), ah1 = ldsA(hbuf + 32, H_, lane),
         ah2 = ldsA(hbuf + 64, H_, lane), ah3 = ldsA(hbuf + 96, H_, lane);
    v8f loc[4];
#pragma unroll
    for (int nt = 0; nt < 4; ++nt) {
      v8f g = bias8(bl + BGo + nt*16, lane);
      g = wmma_f16(ah0, ldB(Wl + OFF_Wg + (nt*4+0)*512, lane), g);
      g = wmma_f16(ah1, ldB(Wl + OFF_Wg + (nt*4+1)*512, lane), g);
      g = wmma_f16(ah2, ldB(Wl + OFF_Wg + (nt*4+2)*512, lane), g);
      g = wmma_f16(ah3, ldB(Wl + OFF_Wg + (nt*4+3)*512, lane), g);
      v8f p = bias8(bl + BPo + nt*16, lane);
      p = wmma_f16(ah0, ldB(Wl + OFF_Wp + (nt*4+0)*512, lane), p);
      p = wmma_f16(ah1, ldB(Wl + OFF_Wp + (nt*4+1)*512, lane), p);
      p = wmma_f16(ah2, ldB(Wl + OFF_Wp + (nt*4+2)*512, lane), p);
      p = wmma_f16(ah3, ldB(Wl + OFF_Wp + (nt*4+3)*512, lane), p);
      v8f zz = bias8(bl + BZo + nt*16, lane);
      zz = wmma_f16(az0, ldB(Wl + OFF_Wz + (nt*2+0)*512, lane), zz);
      zz = wmma_f16(az1, ldB(Wl + OFF_Wz + (nt*2+1)*512, lane), zz);
#pragma unroll
      for (int r = 0; r < 8; ++r) {
        float gg = sigf(g[r]);
        loc[nt][r] = (1.f - gg)*zz[r] + gg*p[r];
      }
      storeD(pbuf + nt*16, Z_, lane, p, true);
    }

    // ---- scale = softplus(relu(prop)@Wsc + bsc)+1e-3 ; z = loc + scale*eps ; base_lp
    v16h ap0 = ldsA(pbuf, Z_, lane), ap1 = ldsA(pbuf + 32, Z_, lane);
    v8f zD[4]; v8f lp;
#pragma unroll
    for (int r = 0; r < 8; ++r) lp[r] = 0.f;
#pragma unroll
    for (int nt = 0; nt < 4; ++nt) {
      v8f sc = bias8(bl + BSCo + nt*16, lane);
      sc = wmma_f16(ap0, ldB(Wl + OFF_Wsc + (nt*2+0)*512, lane), sc);
      sc = wmma_f16(ap1, ldB(Wl + OFF_Wsc + (nt*2+1)*512, lane), sc);
      const float* ep = eps + ((size_t)t*N_ + rowBase + mb)*Z_ + nt*16 + nD;
#pragma unroll
      for (int r = 0; r < 8; ++r) {
        float s  = softplusf(sc[r]) + 0.001f;
        float e_ = ep[(size_t)r*Z_];
        zD[nt][r] = loc[nt][r] + s*e_;
        lp[r] += -0.5f*e_*e_ - __logf(s) - 0.5f*LOG2PI_;
      }
      storeD(zbuf + nt*16, Z_, lane, zD[nt], false);
    }

    // ---- IAF flows
#pragma unroll
    for (int f = 0; f < NF_; ++f) {
      v16h za0 = ldsA(zbuf, Z_, lane), za1 = ldsA(zbuf + 32, Z_, lane);
#pragma unroll
      for (int nt = 0; nt < 8; ++nt) {
        v8f acc = bias8(bl + B1o + f*128 + nt*16, lane);
        acc = wmma_f16(za0, ldB(Wl + OFF_W1 + f*8192 + (nt*2+0)*512, lane), acc);
        acc = wmma_f16(za1, ldB(Wl + OFF_W1 + f*8192 + (nt*2+1)*512, lane), acc);
        storeD(hbuf + nt*16, H_, lane, acc, true);
      }
      v16h hh0 = ldsA(hbuf,      H_, lane), hh1 = ldsA(hbuf + 32, H_, lane),
           hh2 = ldsA(hbuf + 64, H_, lane), hh3 = ldsA(hbuf + 96, H_, lane);
#pragma unroll
      for (int nt = 0; nt < 4; ++nt) {
        v8f mu = bias8(bl + BMo + f*64 + nt*16, lane);
        mu = wmma_f16(hh0, ldB(Wl + OFF_Wm + f*8192 + (nt*4+0)*512, lane), mu);
        mu = wmma_f16(hh1, ldB(Wl + OFF_Wm + f*8192 + (nt*4+1)*512, lane), mu);
        mu = wmma_f16(hh2, ldB(Wl + OFF_Wm + f*8192 + (nt*4+2)*512, lane), mu);
        mu = wmma_f16(hh3, ldB(Wl + OFF_Wm + f*8192 + (nt*4+3)*512, lane), mu);
        v8f sp = bias8(bl + BSo + f*64 + nt*16, lane);
        sp = wmma_f16(hh0, ldB(Wl + OFF_Ws + f*8192 + (nt*4+0)*512, lane), sp);
        sp = wmma_f16(hh1, ldB(Wl + OFF_Ws + f*8192 + (nt*4+1)*512, lane), sp);
        sp = wmma_f16(hh2, ldB(Wl + OFF_Ws + f*8192 + (nt*4+2)*512, lane), sp);
        sp = wmma_f16(hh3, ldB(Wl + OFF_Ws + f*8192 + (nt*4+3)*512, lane), sp);
#pragma unroll
        for (int r = 0; r < 8; ++r) {
          float sg = sigf(sp[r] + 1.0f);
          zD[nt][r] = sg*zD[nt][r] + (1.f - sg)*mu[r];
          lp[r] -= __logf(sg);
        }
        storeD(zbuf + nt*16, Z_, lane, zD[nt], false);
      }
    }

    // ---- z_logprob: reduce row-sum over 16 lanes of each half
#pragma unroll
    for (int r = 0; r < 8; ++r) {
      float v = lp[r];
      v += __shfl_xor(v, 1, 32);
      v += __shfl_xor(v, 2, 32);
      v += __shfl_xor(v, 4, 32);
      v += __shfl_xor(v, 8, 32);
      if ((lane & 15) == 0) o_zlp[(size_t)t*N_ + rowBase + mb + r] = v;
    }

    // ---- emission: e = relu(z@We1+be1)@We2+be2 ; sample + logprobs + fraction
    v16h ze0 = ldsA(zbuf, Z_, lane), ze1 = ldsA(zbuf + 32, Z_, lane);
#pragma unroll
    for (int nt = 0; nt < 8; ++nt) {
      v8f acc = bias8(bl + BE1o + nt*16, lane);
      acc = wmma_f16(ze0, ldB(Wl + OFF_We1 + (nt*2+0)*512, lane), acc);
      acc = wmma_f16(ze1, ldB(Wl + OFF_We1 + (nt*2+1)*512, lane), acc);
      storeD(hbuf + nt*16, H_, lane, acc, true);
    }
    v16h et0 = ldsA(hbuf,      H_, lane), et1 = ldsA(hbuf + 32, H_, lane),
         et2 = ldsA(hbuf + 64, H_, lane), et3 = ldsA(hbuf + 96, H_, lane);
#pragma unroll
    for (int nt = 0; nt < 2; ++nt) {
      v8f e = bias8(bl + BE2o + nt*16, lane);
      e = wmma_f16(et0, ldB(Wl + OFF_We2 + (nt*4+0)*512, lane), e);
      e = wmma_f16(et1, ldB(Wl + OFF_We2 + (nt*4+1)*512, lane), e);
      e = wmma_f16(et2, ldB(Wl + OFF_We2 + (nt*4+2)*512, lane), e);
      e = wmma_f16(et3, ldB(Wl + OFF_We2 + (nt*4+3)*512, lane), e);
      const int j  = nt*16 + nD;            // column of e (0..31) = a*NC+c
      const int n0 = rowBase + mb;
      if (j < 3*NC_) {                      // logits channels a=0..2
        const int a = j >> 3, c = j & 7;
        const size_t base = (((size_t)t*3 + a)*NC_ + c)*N_ + n0;
#pragma unroll
        for (int r = 0; r < 8; ++r) {
          float xv  = e[r];
          float sg  = sigf(xv);
          float smp = (u[base + r] < sg) ? 1.f : 0.f;
          float lsp = -softplusf(-xv);      // log_sigmoid(x)
          float lsn = -softplusf(xv);       // log_sigmoid(-x)
          o_samp[base + r] = smp;
          o_xlp[base + r]  = smp*lsp + (1.f - smp)*lsn;
        }
      } else {                              // fraction channel
        const int c = j & 7;
        const size_t base = ((size_t)t*NC_ + c)*N_ + n0;
#pragma unroll
        for (int r = 0; r < 8; ++r) o_frac[base + r] = sigf(e[r]);
      }
    }
  }
}

// ---------------- host launch ----------------
extern "C" void kernel_launch(void* const* d_in, const int* in_sizes, int n_in,
                              void* d_out, int out_size, void* d_ws, size_t ws_size,
                              hipStream_t stream) {
  (void)in_sizes; (void)n_in; (void)out_size; (void)ws_size;
  const float* x   = (const float*)d_in[0];
  const float* eps = (const float*)d_in[1];
  const float* u   = (const float*)d_in[2];
  _Float16* pw = (_Float16*)d_ws;    // needs 196,608 B of workspace

  hipFuncSetAttribute((const void*)ssm_kernel,
                      hipFuncAttributeMaxDynamicSharedMemorySize, SMEM_BYTES);

  // pack weights (mask pre-applied for IAF) into WMMA-B fragment order
  pack_weights<<< 8, 512, 0, stream>>>((const float*)d_in[3],  32, 128, pw + OFF_Wxh, 0);
  pack_weights<<<16, 512, 0, stream>>>((const float*)d_in[4],  64, 128, pw + OFF_Wzh, 0);
  pack_weights<<<16, 512, 0, stream>>>((const float*)d_in[6], 128,  64, pw + OFF_Wg,  0);
  pack_weights<<<16, 512, 0, stream>>>((const float*)d_in[8], 128,  64, pw + OFF_Wp,  0);
  pack_weights<<< 8, 512, 0, stream>>>((const float*)d_in[10], 64,  64, pw + OFF_Wz,  0);
  pack_weights<<< 8, 512, 0, stream>>>((const float*)d_in[12], 64,  64, pw + OFF_Wsc, 0);
  for (int f = 0; f < NF_; ++f) {
    pack_weights<<<16, 512, 0, stream>>>((const float*)d_in[14] + f*Z_*HI_,  64, 128, pw + OFF_W1 + f*8192, 1);
    pack_weights<<<16, 512, 0, stream>>>((const float*)d_in[16] + f*HI_*Z_, 128,  64, pw + OFF_Wm + f*8192, 2);
    pack_weights<<<16, 512, 0, stream>>>((const float*)d_in[18] + f*HI_*Z_, 128,  64, pw + OFF_Ws + f*8192, 2);
  }
  pack_weights<<<16, 512, 0, stream>>>((const float*)d_in[20], 64, 128, pw + OFF_We1, 0);
  pack_weights<<< 8, 512, 0, stream>>>((const float*)d_in[22],128,  32, pw + OFF_We2, 0);

  float* o = (float*)d_out;
  float* o_samp = o;
  float* o_frac = o_samp + (size_t)S_*3*NC_*N_;
  float* o_xlp  = o_frac + (size_t)S_*NC_*N_;
  float* o_zlp  = o_xlp  + (size_t)S_*3*NC_*N_;

  ssm_kernel<<<N_/(16*WAVES), WAVES*32, SMEM_BYTES, stream>>>(
      x, eps, u, pw,
      (const float*)d_in[5],  (const float*)d_in[7],  (const float*)d_in[9],
      (const float*)d_in[11], (const float*)d_in[13], (const float*)d_in[15],
      (const float*)d_in[17], (const float*)d_in[19], (const float*)d_in[21],
      (const float*)d_in[23],
      o_samp, o_frac, o_xlp, o_zlp);
}
